// GatedDeltaNet_57964878627186
// MI455X (gfx1250) — matrix operand order repaired and checked by reading, not compiled, not run on hardware
//
#include <hip/hip_runtime.h>
#include <math.h>

#define B_      2
#define L_      4096
#define HID_    1024
#define NH_     12
#define DK_     64
#define DV_     128
#define KD_     768
#define VD_     1536
#define NCOL_   4608            // KD + KD + VD + VD
#define M_      (B_ * L_)       // 8192
#define CHUNK_  64
#define NCHUNK_ (L_ / CHUNK_)   // 64

typedef __attribute__((ext_vector_type(16))) __bf16 v16bf;
typedef __attribute__((ext_vector_type(8)))  float  v8f;

#define WMMA_BF16(a, b, c) \
  __builtin_amdgcn_wmma_f32_16x16x32_bf16(false, (a), false, (b), (short)0, (c), false, false)

static __device__ __forceinline__ unsigned short f2bf(float f) {
  unsigned u = __builtin_bit_cast(unsigned, f);
  u += 0x7FFFu + ((u >> 16) & 1u);   // round-to-nearest-even
  return (unsigned short)(u >> 16);
}
static __device__ __forceinline__ float sigmoidf_(float x) { return 1.f / (1.f + __expf(-x)); }
static __device__ __forceinline__ float siluf_(float x)    { return x * sigmoidf_(x); }

// A operand: lane m = lane&15 holds row m; elems 0-7 = K[hi*8..], elems 8-15 = K[16+hi*8..]
static __device__ __forceinline__ v16bf load_a_frag(const unsigned short* tile, int ld,
                                                    int row0, int k0, int lane) {
  int r = row0 + (lane & 15), hi = lane >> 4;
  const __bf16* p0 = (const __bf16*)(tile + (size_t)r * ld + k0 + hi * 8);
  const __bf16* p1 = (const __bf16*)(tile + (size_t)r * ld + k0 + 16 + hi * 8);
  v16bf v;
#pragma unroll
  for (int i = 0; i < 8; ++i) { v[i] = p0[i]; v[8 + i] = p1[i]; }
  return v;
}
// B operand: lane n = lane&15 holds column n; elems 0-7 = K[hi*8..], elems 8-15 = K[16+hi*8..]
// Source tile must be stored transposed: tile[n][k].
static __device__ __forceinline__ v16bf load_b_frag(const unsigned short* tile, int ld,
                                                    int col0, int k0, int lane) {
  int r = col0 + (lane & 15), hi = lane >> 4;
  const __bf16* p0 = (const __bf16*)(tile + (size_t)r * ld + k0 + hi * 8);
  const __bf16* p1 = (const __bf16*)(tile + (size_t)r * ld + k0 + 16 + hi * 8);
  v16bf v;
#pragma unroll
  for (int i = 0; i < 8; ++i) { v[i] = p0[i]; v[8 + i] = p1[i]; }
  return v;
}
static __device__ __forceinline__ v8f zero8() {
  v8f z;
#pragma unroll
  for (int i = 0; i < 8; ++i) z[i] = 0.f;
  return z;
}
// C/D: elem r <-> (M = row0 + r + 8*hi, N = col0 + (lane&15))
static __device__ __forceinline__ void store_c_lds(float* dst, int ld, int row0, int col0,
                                                   int lane, v8f c) {
  int col = col0 + (lane & 15);
  int rb  = row0 + 8 * (lane >> 4);
#pragma unroll
  for (int r = 0; r < 8; ++r) dst[(rb + r) * ld + col] = c[r];
}

// ---- CDNA5 async global->LDS copy (ASYNCcnt-tracked) ----------------------
static __device__ __forceinline__ void async_ld_b128(unsigned lds_off, const void* gptr) {
  asm volatile("global_load_async_to_lds_b128 %0, %1, off"
               :: "v"(lds_off), "v"(gptr) : "memory");
}
static __device__ __forceinline__ void wait_async0() {
  asm volatile("s_wait_asynccnt 0x0" ::: "memory");
}

// ---------------------------------------------------------------------------
// Elementwise conversion kernels
// ---------------------------------------------------------------------------
__global__ void k_cvt_bf16(const float* __restrict__ s, unsigned short* __restrict__ d, int n) {
  int i = blockIdx.x * 256 + threadIdx.x;
  if (i < n) d[i] = f2bf(s[i]);
}
// W [K][N] f32 -> WT [N][K] bf16
__global__ void k_wtrans(const float* __restrict__ W, unsigned short* __restrict__ WT,
                         int K, int N) {
  int i = blockIdx.x * 256 + threadIdx.x;
  if (i < K * N) {
    int k = i / N, n = i - k * N;
    WT[(size_t)n * K + k] = f2bf(W[i]);
  }
}

// ---------------------------------------------------------------------------
// bf16 WMMA GEMM: C[M][N] f32 = A[M][K] bf16 @ Bt[N][K] bf16.
// Block 256 thr = 8 waves as 4(M)x2(N); wave tile 32x64; block tile 128x128.
// Weight tile double-buffered in LDS via async load-to-LDS.
// ---------------------------------------------------------------------------
static __device__ __forceinline__ void fill_btile(unsigned short* dst, const unsigned short* Bt,
                                                  int bn0, int K, int k0, int tid) {
#pragma unroll
  for (int u = 0; u < 2; ++u) {
    int id  = tid + 256 * u;        // 512 x 16B transfers = 128 rows x 32 k
    int row = id >> 2, kc = id & 3;
    unsigned lds_off = (unsigned)(size_t)(dst + row * 32 + kc * 8);
    const void* g = Bt + (size_t)(bn0 + row) * K + k0 + kc * 8;
    async_ld_b128(lds_off, g);
  }
}

__global__ __launch_bounds__(256) void k_gemm_bf16(const unsigned short* __restrict__ A,
                                                   const unsigned short* __restrict__ Bt,
                                                   float* __restrict__ C,
                                                   int M, int N, int K) {
  __shared__ unsigned short bsh[2][128 * 32];   // 16 KB double buffer
  const int tid = threadIdx.x, lane = tid & 31, wid = tid >> 5;
  const int bm  = blockIdx.x * 128 + (wid & 3) * 32;
  const int bn0 = blockIdx.y * 128;
  const int wn  = (wid >> 2) * 64;
  v8f acc[2][4];
#pragma unroll
  for (int i = 0; i < 2; ++i)
#pragma unroll
    for (int j = 0; j < 4; ++j) acc[i][j] = zero8();

  const int KT = K / 32;
  fill_btile(bsh[0], Bt, bn0, K, 0, tid);
  for (int it = 0; it < KT; ++it) {
    wait_async0();          // my async transfers for current buffer done
    __syncthreads();        // everyone's done; prev buffer no longer read
    if (it + 1 < KT) fill_btile(bsh[(it + 1) & 1], Bt, bn0, K, (it + 1) * 32, tid);
    const unsigned short* bs = bsh[it & 1];
    const int k0 = it * 32;
    if (k0 + 32 < K)
      __builtin_prefetch(A + (size_t)(bm + (lane & 15)) * K + k0 + 32);
    // Load all operands into distinct registers first, then stream 8 WMMAs
    // without intervening LDS waits.
    v16bf a0 = load_a_frag(A, K, bm, k0, lane);
    v16bf a1 = load_a_frag(A, K, bm + 16, k0, lane);
    v16bf b0 = load_b_frag(bs, 32, wn + 0,  0, lane);
    v16bf b1 = load_b_frag(bs, 32, wn + 16, 0, lane);
    v16bf b2 = load_b_frag(bs, 32, wn + 32, 0, lane);
    v16bf b3 = load_b_frag(bs, 32, wn + 48, 0, lane);
    acc[0][0] = WMMA_BF16(a0, b0, acc[0][0]);
    acc[1][0] = WMMA_BF16(a1, b0, acc[1][0]);
    acc[0][1] = WMMA_BF16(a0, b1, acc[0][1]);
    acc[1][1] = WMMA_BF16(a1, b1, acc[1][1]);
    acc[0][2] = WMMA_BF16(a0, b2, acc[0][2]);
    acc[1][2] = WMMA_BF16(a1, b2, acc[1][2]);
    acc[0][3] = WMMA_BF16(a0, b3, acc[0][3]);
    acc[1][3] = WMMA_BF16(a1, b3, acc[1][3]);
  }
#pragma unroll
  for (int i = 0; i < 2; ++i)
#pragma unroll
    for (int j = 0; j < 4; ++j) {
      int rb  = bm + i * 16 + 8 * (lane >> 4);
      int col = bn0 + wn + j * 16 + (lane & 15);
#pragma unroll
      for (int r = 0; r < 8; ++r) C[(size_t)(rb + r) * N + col] = acc[i][j][r];
    }
}

// ---------------------------------------------------------------------------
// gk / beta small projections: one block per row m of hidden, 8 waves x 3 cols
// ---------------------------------------------------------------------------
__global__ __launch_bounds__(256) void k_smallproj(const float* __restrict__ hs,
                                                   const float* __restrict__ Wgk,
                                                   const float* __restrict__ Wb,
                                                   const float* __restrict__ b_b,
                                                   const float* __restrict__ A_log,
                                                   const float* __restrict__ dtb,
                                                   float* __restrict__ gkO,
                                                   float* __restrict__ btO) {
  const int m = blockIdx.x;                  // b*L + l
  const int b = m / L_, l = m - b * L_;
  const int wid = threadIdx.x >> 5, lane = threadIdx.x & 31;
  const float* row = hs + (size_t)m * HID_;
#pragma unroll
  for (int j = 0; j < 3; ++j) {
    int o = wid * 3 + j;                     // 0..23
    int h = (o < NH_) ? o : (o - NH_);
    const float* W = (o < NH_) ? Wgk : Wb;
    float acc = 0.f;
    for (int k = lane; k < HID_; k += 32) acc += row[k] * W[k * NH_ + h];
#pragma unroll
    for (int off = 16; off; off >>= 1) acc += __shfl_xor(acc, off, 32);
    if (lane == 0) {
      size_t idx = ((size_t)(b * NH_ + h)) * L_ + l;
      if (o < NH_) {
        float x  = acc + dtb[h];
        float sp = (x > 20.f) ? x : log1pf(__expf(x));
        gkO[idx] = -__expf(A_log[h]) * sp;
      } else {
        btO[idx] = sigmoidf_(acc + b_b[h]);
      }
    }
  }
}

// ---------------------------------------------------------------------------
// Causal depthwise conv(4) + SiLU + per-head l2norm (*scale) for q / k.
// ---------------------------------------------------------------------------
__global__ __launch_bounds__(256) void k_convqk(const float* __restrict__ X, int coff,
                                                const float* __restrict__ w,
                                                float* __restrict__ out, float scale) {
  const int wid = threadIdx.x >> 5, lane = threadIdx.x & 31;
  const int gidx = blockIdx.x * 8 + wid;     // over (b*NH+h)*L + l
  const int l  = gidx % L_;
  const int bh = gidx / L_;
  const int h  = bh % NH_;
  const int b  = bh / NH_;
  float y[2];
  float ss = 0.f;
#pragma unroll
  for (int u = 0; u < 2; ++u) {
    int d  = lane + 32 * u;
    int cc = h * DK_ + d;
    float acc = 0.f;
#pragma unroll
    for (int j = 0; j < 4; ++j) {
      int ll = l - 3 + j;
      if (ll >= 0) acc += X[((size_t)(b * L_ + ll)) * NCOL_ + coff + cc] * w[cc * 4 + j];
    }
    y[u] = siluf_(acc);
    ss += y[u] * y[u];
  }
#pragma unroll
  for (int off = 16; off; off >>= 1) ss += __shfl_xor(ss, off, 32);
  float r = rsqrtf(ss + 1e-6f) * scale;
#pragma unroll
  for (int u = 0; u < 2; ++u)
    out[(((size_t)bh) * L_ + l) * DK_ + lane + 32 * u] = y[u] * r;
}

// Causal depthwise conv(4) + SiLU for v, reorder to [b][h][l][dv]
__global__ __launch_bounds__(256) void k_convv(const float* __restrict__ X,
                                               const float* __restrict__ w,
                                               float* __restrict__ out) {
  int idx = blockIdx.x * 256 + threadIdx.x;  // over B*L*VD
  int c  = idx % VD_;
  int ml = idx / VD_;
  int l  = ml % L_;
  int b  = ml / L_;
  float acc = 0.f;
#pragma unroll
  for (int j = 0; j < 4; ++j) {
    int ll = l - 3 + j;
    if (ll >= 0) acc += X[((size_t)(b * L_ + ll)) * NCOL_ + (2 * KD_) + c] * w[c * 4 + j];
  }
  float s = siluf_(acc);
  int h = c >> 7, dv = c & 127;
  out[(((size_t)(b * NH_ + h)) * L_ + l) * DV_ + dv] = s;
}

// ---------------------------------------------------------------------------
// Chunked gated delta rule scan. One block per (b,h); 64 chunks of 64 steps.
// ---------------------------------------------------------------------------
#define SC_S    0
#define SC_KK   (SC_S    + 64 * 128 * 4)
#define SC_QK   (SC_KK   + 64 * 64 * 4)
#define SC_QBF  (SC_QK   + 64 * 64 * 4)
#define SC_KBF  (SC_QBF  + 64 * 64 * 2)
#define SC_KEBF (SC_KBF  + 64 * 64 * 2)
#define SC_QEBF (SC_KEBF + 64 * 64 * 2)
#define SC_K2T  (SC_QEBF + 64 * 64 * 2)
#define SC_TBF  (SC_K2T  + 64 * 64 * 2)
#define SC_W2BF (SC_TBF  + 64 * 64 * 2)
#define SC_RHST (SC_W2BF + 64 * 64 * 2)
#define SC_DT   (SC_RHST + 128 * 64 * 2)
#define SC_S0T  (SC_DT   + 128 * 64 * 2)
#define SC_SCAL (SC_S0T  + 128 * 64 * 2)
#define SCAN_LDS_BYTES (SC_SCAL + 4 * 64 * 4)   // 173,056 bytes < 320 KB

__global__ __launch_bounds__(256) void k_scan(const float* __restrict__ qh,
                                              const float* __restrict__ kh,
                                              const float* __restrict__ vh,
                                              const float* __restrict__ gkp,
                                              const float* __restrict__ btp,
                                              float* __restrict__ oh) {
  extern __shared__ char smem[];
  float* S   = (float*)(smem + SC_S);      // [64 dk][128 dv] persistent state
  float* KKf = (float*)(smem + SC_KK);     // KK^T, then masked A
  float* QKf = (float*)(smem + SC_QK);     // QK^T, then aliased as X (T solve)
  float* Xf  = QKf;
  unsigned short* qbf  = (unsigned short*)(smem + SC_QBF);   // [t][d]
  unsigned short* kbf  = (unsigned short*)(smem + SC_KBF);   // [t][d]
  unsigned short* kebf = (unsigned short*)(smem + SC_KEBF);  // k*e^{G_t}
  unsigned short* qebf = (unsigned short*)(smem + SC_QEBF);  // q*e^{G_t}
  unsigned short* k2T  = (unsigned short*)(smem + SC_K2T);   // [d][s] k_s*e^{G_C-G_s}
  unsigned short* Tbf  = (unsigned short*)(smem + SC_TBF);   // (I+A)^-1 (also scratch)
  unsigned short* w2bf = (unsigned short*)(smem + SC_W2BF);  // tril(QK*decay)
  unsigned short* rhsT = (unsigned short*)(smem + SC_RHST);  // [e][s] rhs^T
  unsigned short* dT   = (unsigned short*)(smem + SC_DT);    // [e][s] Delta^T
  unsigned short* s0T  = (unsigned short*)(smem + SC_S0T);   // [e][d] S0^T
  float* gbuf = (float*)(smem + SC_SCAL);
  float* bbuf = gbuf + 64;
  float* cum  = gbuf + 128;
  float* eg   = gbuf + 192;
  // scratch tiles for the block-inverse combine (alias Tbf: 4 x 32x32 = 8 KB)
  unsigned short* a21bf = Tbf;
  unsigned short* v11T  = Tbf + 1024;
  unsigned short* v22bf = Tbf + 2048;
  unsigned short* tmpT  = Tbf + 3072;

  const int tid = threadIdx.x, lane = tid & 31, wid = tid >> 5;
  const int bh = blockIdx.x;
  const float* qG  = qh  + (size_t)bh * L_ * DK_;
  const float* kG  = kh  + (size_t)bh * L_ * DK_;
  const float* vG  = vh  + (size_t)bh * L_ * DV_;
  const float* gkG = gkp + (size_t)bh * L_;
  const float* btG = btp + (size_t)bh * L_;
  float* oG = oh + (size_t)bh * L_ * DV_;

  for (int i = tid; i < 64 * 128; i += 256) S[i] = 0.f;
  __syncthreads();

  for (int c = 0; c < NCHUNK_; ++c) {
    const int l0 = c * CHUNK_;
    if (tid < 64) { gbuf[tid] = gkG[l0 + tid]; bbuf[tid] = btG[l0 + tid]; }
    __syncthreads();
    if (tid == 0) { float a = 0.f; for (int t = 0; t < 64; ++t) { a += gbuf[t]; cum[t] = a; } }
    __syncthreads();
    if (tid < 64) eg[tid] = __expf(cum[tid]);
    __syncthreads();
    const float cumC = cum[63];

    // --- stage tiles into LDS (f32 -> bf16) ---
    for (int i = tid; i < 64 * 64; i += 256) {
      int t = i >> 6, d = i & 63;
      float qv = qG[(size_t)(l0 + t) * DK_ + d];
      float kv = kG[(size_t)(l0 + t) * DK_ + d];
      qbf[i]  = f2bf(qv);
      kbf[i]  = f2bf(kv);
      kebf[i] = f2bf(kv * eg[t]);
      qebf[i] = f2bf(qv * eg[t]);
      k2T[d * 64 + t] = f2bf(kv * __expf(cumC - cum[t]));
    }
    for (int i = tid; i < 128 * 64; i += 256) {
      int e = i >> 6, d = i & 63;
      s0T[i] = f2bf(S[d * 128 + e]);
    }
    __syncthreads();

    // --- stage 1: KK^T (16 frags), QK^T (16), (Ke@S0)->rhs (32) ---
    for (int f = wid; f < 64; f += 8) {
      if (f < 16) {
        int mi = f >> 2, ni = f & 3;
        v8f acc = zero8();
#pragma unroll
        for (int kk = 0; kk < 64; kk += 32)
          acc = WMMA_BF16(load_a_frag(kbf, 64, mi * 16, kk, lane),
                          load_b_frag(kbf, 64, ni * 16, kk, lane), acc);
        store_c_lds(KKf, 64, mi * 16, ni * 16, lane, acc);
      } else if (f < 32) {
        int g = f - 16, mi = g >> 2, ni = g & 3;
        v8f acc = zero8();
#pragma unroll
        for (int kk = 0; kk < 64; kk += 32)
          acc = WMMA_BF16(load_a_frag(qbf, 64, mi * 16, kk, lane),
                          load_b_frag(kbf, 64, ni * 16, kk, lane), acc);
        store_c_lds(QKf, 64, mi * 16, ni * 16, lane, acc);
      } else {
        int g = f - 32, mi = g >> 3, ni = g & 7;
        v8f acc = zero8();
#pragma unroll
        for (int kk = 0; kk < 64; kk += 32)
          acc = WMMA_BF16(load_a_frag(kebf, 64, mi * 16, kk, lane),
                          load_b_frag(s0T, 64, ni * 16, kk, lane), acc);
        int col = ni * 16 + (lane & 15), rb = mi * 16 + 8 * (lane >> 4);
#pragma unroll
        for (int r = 0; r < 8; ++r) {
          int t = rb + r;
          float val = bbuf[t] * (vG[(size_t)(l0 + t) * DV_ + col] - acc[r]);
          rhsT[col * 64 + t] = f2bf(val);
        }
      }
    }
    __syncthreads();

    // --- stage 2: mask/scale A; build w2bf = tril(QK * decay) ---
    for (int i = tid; i < 64 * 64; i += 256) {
      int t = i >> 6, s = i & 63;
      float dec = __expf(cum[t] - cum[s]);
      w2bf[i] = f2bf((s <= t) ? dec * QKf[i] : 0.f);
      KKf[i]  = (s < t) ? bbuf[t] * dec * KKf[i] : 0.f;
    }
    __syncthreads();

    // --- T = (I+A)^-1 via block-triangular inverse ---
    // X = I; invert both 32x32 diagonal blocks concurrently (31 steps)
    for (int i = tid; i < 64 * 64; i += 256) Xf[i] = ((i >> 6) == (i & 63)) ? 1.f : 0.f;
    __syncthreads();
    for (int s = 0; s < 31; ++s) {
      int nj = s + 1, tot = (31 - s) * nj;
      for (int e = tid; e < 2 * tot; e += 256) {
        int blk = (e >= tot) ? 1 : 0;
        int idx = e - blk * tot;
        int t = s + 1 + idx / nj, j = idx - (idx / nj) * nj;
        int o = blk * 32;
        Xf[(o + t) * 64 + (o + j)] -= KKf[(o + t) * 64 + (o + s)] * Xf[(o + s) * 64 + (o + j)];
      }
      __syncthreads();
    }
    // stage combine tiles: A21, V11^T, V22 as bf16 (scratch aliases Tbf)
    for (int i = tid; i < 32 * 32; i += 256) {
      int r = i >> 5, cc = i & 31;
      a21bf[i]           = f2bf(KKf[(32 + r) * 64 + cc]);
      v11T[cc * 32 + r]  = f2bf(Xf[r * 64 + cc]);
      v22bf[i]           = f2bf(Xf[(32 + r) * 64 + 32 + cc]);
    }
    __syncthreads();
    // tmp = A21 @ V11  (32x32, K=32) -> tmpT
    if (wid < 4) {
      int mi = wid >> 1, ni = wid & 1;
      v8f acc = WMMA_BF16(load_a_frag(a21bf, 32, mi * 16, 0, lane),
                          load_b_frag(v11T, 32, ni * 16, 0, lane), zero8());
      int col = ni * 16 + (lane & 15), rb = mi * 16 + 8 * (lane >> 4);
#pragma unroll
      for (int r = 0; r < 8; ++r) tmpT[col * 32 + rb + r] = f2bf(acc[r]);
    }
    __syncthreads();
    // V21 = -V22 @ tmp -> lower-left block of X
    if (wid < 4) {
      int mi = wid >> 1, ni = wid & 1;
      v8f acc = WMMA_BF16(load_a_frag(v22bf, 32, mi * 16, 0, lane),
                          load_b_frag(tmpT, 32, ni * 16, 0, lane), zero8());
      int col = ni * 16 + (lane & 15), rb = mi * 16 + 8 * (lane >> 4);
#pragma unroll
      for (int r = 0; r < 8; ++r) Xf[(32 + rb + r) * 64 + col] = -acc[r];
    }
    __syncthreads();
    for (int i = tid; i < 64 * 64; i += 256) Tbf[i] = f2bf(Xf[i]);
    __syncthreads();

    // --- stage 3: Delta = T @ rhs -> dT (transposed bf16) ---
    for (int f = wid; f < 32; f += 8) {
      int mi = f >> 3, ni = f & 7;
      v8f acc = zero8();
#pragma unroll
      for (int kk = 0; kk < 64; kk += 32)
        acc = WMMA_BF16(load_a_frag(Tbf, 64, mi * 16, kk, lane),
                        load_b_frag(rhsT, 64, ni * 16, kk, lane), acc);
      int col = ni * 16 + (lane & 15), rb = mi * 16 + 8 * (lane >> 4);
#pragma unroll
      for (int r = 0; r < 8; ++r) dT[col * 64 + rb + r] = f2bf(acc[r]);
    }
    __syncthreads();

    // --- stage 4+5: O = Qe@S0 + W2@Delta ; S = egC*S + K2^T@Delta ---
    const float egC = eg[63];
    for (int f = wid; f < 64; f += 8) {
      if (f < 32) {
        int mi = f >> 3, ni = f & 7;
        v8f acc = zero8();
#pragma unroll
        for (int kk = 0; kk < 64; kk += 32)
          acc = WMMA_BF16(load_a_frag(qebf, 64, mi * 16, kk, lane),
                          load_b_frag(s0T, 64, ni * 16, kk, lane), acc);
#pragma unroll
        for (int kk = 0; kk < 64; kk += 32)
          acc = WMMA_BF16(load_a_frag(w2bf, 64, mi * 16, kk, lane),
                          load_b_frag(dT, 64, ni * 16, kk, lane), acc);
        int col = ni * 16 + (lane & 15), rb = mi * 16 + 8 * (lane >> 4);
#pragma unroll
        for (int r = 0; r < 8; ++r)
          oG[(size_t)(l0 + rb + r) * DV_ + col] = acc[r];
      } else {
        int g = f - 32, mi = g >> 3, ni = g & 7;
        int col = ni * 16 + (lane & 15), rb = mi * 16 + 8 * (lane >> 4);
        v8f acc;
#pragma unroll
        for (int r = 0; r < 8; ++r) acc[r] = egC * S[(rb + r) * 128 + col];
#pragma unroll
        for (int kk = 0; kk < 64; kk += 32)
          acc = WMMA_BF16(load_a_frag(k2T, 64, mi * 16, kk, lane),
                          load_b_frag(dT, 64, ni * 16, kk, lane), acc);
#pragma unroll
        for (int r = 0; r < 8; ++r) S[(rb + r) * 128 + col] = acc[r];
      }
    }
    __syncthreads();
  }
}

// ---------------------------------------------------------------------------
// RMSNorm + swish gate -> bf16 for the final GEMM. One block per (b,l,h).
// ---------------------------------------------------------------------------
__global__ __launch_bounds__(128) void k_gate(const float* __restrict__ oh,
                                              const float* __restrict__ X,
                                              const float* __restrict__ gnw,
                                              unsigned short* __restrict__ xo) {
  __shared__ float red[4];
  int h = blockIdx.x % NH_;
  int m = blockIdx.x / NH_;
  int b = m / L_, l = m % L_;
  int dv = threadIdx.x, lane = threadIdx.x & 31, wid = threadIdx.x >> 5;
  float o = oh[(((size_t)(b * NH_ + h)) * L_ + l) * DV_ + dv];
  float ss = o * o;
#pragma unroll
  for (int off = 16; off; off >>= 1) ss += __shfl_xor(ss, off, 32);
  if (lane == 0) red[wid] = ss;
  __syncthreads();
  float var = (red[0] + red[1] + red[2] + red[3]) * (1.f / DV_);
  float g = X[(size_t)m * NCOL_ + (2 * KD_ + VD_) + h * DV_ + dv];
  float val = o * rsqrtf(var + 1e-5f) * gnw[dv] * g * sigmoidf_(g);
  xo[(size_t)m * VD_ + h * DV_ + dv] = f2bf(val);
}

// ---------------------------------------------------------------------------
extern "C" void kernel_launch(void* const* d_in, const int* in_sizes, int n_in,
                              void* d_out, int out_size, void* d_ws, size_t ws_size,
                              hipStream_t stream) {
  const float* hs    = (const float*)d_in[0];
  const float* Wq    = (const float*)d_in[1];
  const float* Wk    = (const float*)d_in[2];
  const float* Wv    = (const float*)d_in[3];
  const float* Wg    = (const float*)d_in[4];
  const float* Wo    = (const float*)d_in[5];
  const float* cq    = (const float*)d_in[6];
  const float* ck    = (const float*)d_in[7];
  const float* cv    = (const float*)d_in[8];
  const float* Wgk   = (const float*)d_in[9];
  const float* Wb    = (const float*)d_in[10];
  const float* b_b   = (const float*)d_in[11];
  const float* A_log = (const float*)d_in[12];
  const float* dtb   = (const float*)d_in[13];
  const float* gnw   = (const float*)d_in[14];
  (void)in_sizes; (void)n_in; (void)out_size; (void)ws_size;

  char* p = (char*)d_ws;
  auto take = [&](size_t bytes) { char* r = p; p += (bytes + 255) & ~(size_t)255; return r; };
  unsigned short* hbf  = (unsigned short*)take((size_t)M_ * HID_ * 2);
  unsigned short* wcat = (unsigned short*)take((size_t)NCOL_ * HID_ * 2);
  unsigned short* wot  = (unsigned short*)take((size_t)HID_ * VD_ * 2);
  float* qkvg = (float*)take((size_t)M_ * NCOL_ * 4);
  float* qh_  = (float*)take((size_t)B_ * NH_ * L_ * DK_ * 4);
  float* kh_  = (float*)take((size_t)B_ * NH_ * L_ * DK_ * 4);
  float* vh_  = (float*)take((size_t)B_ * NH_ * L_ * DV_ * 4);
  float* gkb  = (float*)take((size_t)B_ * NH_ * L_ * 4);
  float* btb  = (float*)take((size_t)B_ * NH_ * L_ * 4);
  float* oh_  = (float*)take((size_t)B_ * NH_ * L_ * DV_ * 4);
  unsigned short* xo = (unsigned short*)take((size_t)M_ * VD_ * 2);

  k_cvt_bf16<<<(M_ * HID_ + 255) / 256, 256, 0, stream>>>(hs, hbf, M_ * HID_);
  k_wtrans<<<(HID_ * KD_ + 255) / 256, 256, 0, stream>>>(Wq, wcat, HID_, KD_);
  k_wtrans<<<(HID_ * KD_ + 255) / 256, 256, 0, stream>>>(Wk, wcat + (size_t)KD_ * HID_, HID_, KD_);
  k_wtrans<<<(HID_ * VD_ + 255) / 256, 256, 0, stream>>>(Wv, wcat + (size_t)(2 * KD_) * HID_, HID_, VD_);
  k_wtrans<<<(HID_ * VD_ + 255) / 256, 256, 0, stream>>>(Wg, wcat + (size_t)(2 * KD_ + VD_) * HID_, HID_, VD_);
  k_wtrans<<<(VD_ * HID_ + 255) / 256, 256, 0, stream>>>(Wo, wot, VD_, HID_);

  k_gemm_bf16<<<dim3(M_ / 128, NCOL_ / 128), 256, 0, stream>>>(hbf, wcat, qkvg, M_, NCOL_, HID_);
  k_smallproj<<<M_, 256, 0, stream>>>(hs, Wgk, Wb, b_b, A_log, dtb, gkb, btb);
  k_convqk<<<(B_ * NH_ * L_) / 8, 256, 0, stream>>>(qkvg, 0, cq, qh_, 0.125f);   // DK^-0.5
  k_convqk<<<(B_ * NH_ * L_) / 8, 256, 0, stream>>>(qkvg, KD_, ck, kh_, 1.0f);
  k_convv<<<(B_ * L_ * VD_) / 256, 256, 0, stream>>>(qkvg, cv, vh_);

  static_assert(SCAN_LDS_BYTES <= 320 * 1024, "LDS over WGP limit");
  hipFuncSetAttribute(reinterpret_cast<const void*>(k_scan),
                      hipFuncAttributeMaxDynamicSharedMemorySize, SCAN_LDS_BYTES);
  k_scan<<<B_ * NH_, 256, SCAN_LDS_BYTES, stream>>>(qh_, kh_, vh_, gkb, btb, oh_);

  k_gate<<<M_ * NH_, 128, 0, stream>>>(oh_, qkvg, gnw, xo);
  k_gemm_bf16<<<dim3(M_ / 128, HID_ / 128), 256, 0, stream>>>(xo, wot, (float*)d_out, M_, HID_, VD_);
}
